// NeuronCircuit_60576218743273
// MI455X (gfx1250) — compile-verified, matrix operand order checked by reading, not run on hardware
//
#include <hip/hip_runtime.h>

// ---- problem constants (from reference) ----
#define T_TOK 2048      // B*S
#define DM 1024         // D_MODEL
#define RK 512          // RANK
#define NH 8            // N_HEADS
#define DH 64           // D_HEAD
#define NN 64           // N_COMPRESS == N_EXPAND
#define KC 8            // K_COMPRESS
#define KE 4            // K_EXPAND
#define SEQ 1024
#define BHN 16          // B * NH

typedef __attribute__((ext_vector_type(16))) __bf16 v16bf;
typedef __attribute__((ext_vector_type(8)))  float  v8f;

union FragBF { v16bf v; uint4 q[2]; };

__device__ __forceinline__ unsigned short f2bf(float f) {
  unsigned int u = __float_as_uint(f);
  u += 0x7FFFu + ((u >> 16) & 1u);   // round-to-nearest-even
  return (unsigned short)(u >> 16);
}

__device__ __forceinline__ v8f wmma_bf16(const FragBF& a, const FragBF& b, v8f c) {
  return __builtin_amdgcn_wmma_f32_16x16x32_bf16(false, a.v, false, b.v,
                                                 (short)0, c, false, false);
}

__device__ __forceinline__ void load_frag(FragBF& f, const unsigned short* p) {
  f.q[0] = *(const uint4*)(p);
  f.q[1] = *(const uint4*)(p + 16);
}

// ---------------- cast / transpose kernels ----------------

// in: [P,Q] f32 per batch z  ->  out: [Q,P] bf16 (K-contiguous B^T layout)
__global__ void transpose_cast_kernel(const float* __restrict__ in,
                                      unsigned short* __restrict__ out,
                                      int P, int Q) {
  __shared__ float tile[16][17];
  int nb = blockIdx.z;
  int p0 = blockIdx.x * 16, q0 = blockIdx.y * 16;
  const float* src = in + (long long)nb * P * Q;
  unsigned short* dst = out + (long long)nb * P * Q;
  tile[threadIdx.y][threadIdx.x] =
      src[(long long)(p0 + threadIdx.y) * Q + (q0 + threadIdx.x)];
  __syncthreads();
  dst[(long long)(q0 + threadIdx.y) * P + (p0 + threadIdx.x)] =
      f2bf(tile[threadIdx.x][threadIdx.y]);
}

__global__ void cast_kernel(const float* __restrict__ in,
                            unsigned short* __restrict__ out, int n) {
  int i = blockIdx.x * 256 + threadIdx.x;
  if (i < n) out[i] = f2bf(in[i]);
}

// ---------------- router + grouping ----------------

__global__ void zero_kernel(int* a, int n) {
  int i = blockIdx.x * 64 + threadIdx.x;
  if (i < n) a[i] = 0;
}

// one block (64 threads) per token: scores, top-k, softmax, count
__global__ void router_kernel(const float* __restrict__ input,
                              const float* __restrict__ W, int Din, int k,
                              int* __restrict__ idx, float* __restrict__ wgt,
                              int* __restrict__ counts) {
  __shared__ float xrow[DM];
  __shared__ float sc[NN];
  int t = blockIdx.x;
  for (int d = threadIdx.x; d < Din; d += 64)
    xrow[d] = input[(long long)t * Din + d];
  __syncthreads();
  {
    int n = threadIdx.x;
    const float* w = W + (long long)n * Din;
    float dot = 0.f;
    for (int d = 0; d < Din; ++d) dot += xrow[d] * w[d];
    sc[n] = dot;
  }
  __syncthreads();
  if (threadIdx.x == 0) {
    unsigned long long used = 0ull;
    float wv[KC]; int wi[KC];
    for (int j = 0; j < k; ++j) {
      float best = -3.4e38f; int bi = 0;
      for (int n = 0; n < NN; ++n)
        if (!((used >> n) & 1ull) && sc[n] > best) { best = sc[n]; bi = n; }
      used |= 1ull << bi;
      wi[j] = bi; wv[j] = best;
    }
    float m = wv[0], s = 0.f;
    for (int j = 0; j < k; ++j) { float e = __expf(wv[j] - m); wv[j] = e; s += e; }
    float inv = 1.f / s;
    for (int j = 0; j < k; ++j) {
      idx[t * k + j] = wi[j];
      wgt[t * k + j] = wv[j] * inv;
      atomicAdd(&counts[wi[j]], 1);
    }
  }
}

// single-thread prefix sum + tile table
__global__ void scan_kernel(const int* __restrict__ counts, int* __restrict__ offsets,
                            int* __restrict__ cursors, int* __restrict__ table,
                            int maxTiles) {
  if (threadIdx.x != 0) return;
  int run = 0, pos = 0;
  for (int n = 0; n < NN; ++n) {
    offsets[n] = run; cursors[n] = 0;
    int c = counts[n]; run += c;
    int tiles = (c + 15) >> 4;
    for (int tt = 0; tt < tiles && pos < maxTiles; ++tt)
      table[pos++] = (n << 16) | tt;
  }
  for (; pos < maxTiles; ++pos) table[pos] = -1;
}

__global__ void scatter_kernel(const int* __restrict__ idx, int k,
                               const int* __restrict__ offsets, int* __restrict__ cursors,
                               int* __restrict__ entries, int total) {
  int i = blockIdx.x * 256 + threadIdx.x;
  if (i >= total) return;
  int t = i / k, j = i - t * k;
  int n = idx[t * k + j];
  int pos = atomicAdd(&cursors[n], 1);
  entries[offsets[n] + pos] = (t << 3) | j;   // token*8 + slot
}

// ---------------- MoE grouped GEMM (WMMA bf16, 16x64 tile per wave) ----------------
// A: gathered token rows [T, Kdim] bf16; B: neuron banks [NN, Ncol, Kdim] bf16 (K-contig)
// invalid rows read from `zrow` (a zeroed dummy row) -> no predication in the K loop
__global__ void moe_gemm_kernel(const unsigned short* __restrict__ Abf, int Kdim,
                                const unsigned short* __restrict__ Bbf, int Ncol,
                                const int* __restrict__ table,
                                const int* __restrict__ offsets,
                                const int* __restrict__ counts,
                                const int* __restrict__ entries,
                                const unsigned short* __restrict__ zrow,
                                float* __restrict__ sel, int kSlots) {
  int desc = table[blockIdx.x];
  if (desc < 0) return;
  int n = desc >> 16;
  int tt = desc & 0xFFFF;
  int lane = threadIdx.x;
  __shared__ int ents[16];
  int base = offsets[n];
  int cnt = counts[n];
  if (lane < 16) {
    int r = tt * 16 + lane;
    ents[lane] = (r < cnt) ? entries[base + r] : -1;
  }
  __syncthreads();
  int myEnt = ents[lane & 15];
  const unsigned short* ap =
      (myEnt >= 0) ? (Abf + (long long)(myEnt >> 3) * Kdim) : zrow;
  int col0 = blockIdx.y * 64 + (lane & 15);
  const unsigned short* bp = Bbf + ((long long)n * Ncol + col0) * Kdim;
  const long long bstep = (long long)16 * Kdim;
  int koff = (lane >> 4) << 3;   // K-halves split across lane[4] (ISA A-frag layout)
  v8f acc0 = {}, acc1 = {}, acc2 = {}, acc3 = {};
  for (int k0 = 0; k0 < Kdim; k0 += 32) {
    FragBF a, b0, b1, b2, b3;
    load_frag(a, ap + k0 + koff);
    const unsigned short* bk = bp + k0 + koff;
    load_frag(b0, bk);
    load_frag(b1, bk + bstep);
    load_frag(b2, bk + 2 * bstep);
    load_frag(b3, bk + 3 * bstep);
    acc0 = wmma_bf16(a, b0, acc0);
    acc1 = wmma_bf16(a, b1, acc1);
    acc2 = wmma_bf16(a, b2, acc2);
    acc3 = wmma_bf16(a, b3, acc3);
  }
  // D layout: col = lane&15; rows v (+8 for upper lane half)
  int rbase = (lane >> 4) << 3;
  for (int v = 0; v < 8; ++v) {
    int e = ents[rbase + v];
    if (e >= 0) {
      int t = e >> 3, j = e & 7;
      float* dst = sel + ((long long)t * kSlots + j) * Ncol + col0;
      dst[0]  = acc0[v];
      dst[16] = acc1[v];
      dst[32] = acc2[v];
      dst[48] = acc3[v];
    }
  }
}

// weighted combine of top-k partials; optional f32 / bf16 / V-transposed bf16 outputs
__global__ void combine_kernel(const float* __restrict__ sel, const float* __restrict__ wgt,
                               int k, int Ncol, float* __restrict__ outF,
                               unsigned short* __restrict__ outB, int mode) {
  int t = blockIdx.x;
  int col = blockIdx.y * 256 + threadIdx.x;
  float s = 0.f;
  for (int j = 0; j < k; ++j)
    s += wgt[t * k + j] * sel[((long long)t * k + j) * Ncol + col];
  if (outF) outF[(long long)t * Ncol + col] = s;
  if (outB) {
    if (mode == 2) {  // V transposed: Vt[(b*NH+h)][d][s]
      int b = t >> 10, ss = t & 1023, h = col >> 6, d = col & 63;
      outB[(((long long)(b * NH + h) * DH) + d) * SEQ + ss] = f2bf(s);
    } else {
      outB[(long long)t * Ncol + col] = f2bf(s);
    }
  }
}

// ---------------- dense batched GEMM (attention), 16x64 tile per wave ----------------
// C[m,n] = scale * sum_k A[m,k]*B[n,k]; per-batch offset = (bh>>3)*outer + (bh&7)*inner
__global__ void gemm_bf16_kernel(const unsigned short* __restrict__ A,
                                 long long aOuter, long long aInner, int lda,
                                 const unsigned short* __restrict__ B,
                                 long long bOuter, long long bInner, int ldb,
                                 float* __restrict__ C,
                                 long long cOuter, long long cInner, int ldc,
                                 int Kdim, float scale) {
  int lane = threadIdx.x;
  int batch = blockIdx.z;
  long long aoff = (long long)(batch >> 3) * aOuter + (long long)(batch & 7) * aInner;
  long long boff = (long long)(batch >> 3) * bOuter + (long long)(batch & 7) * bInner;
  long long coff = (long long)(batch >> 3) * cOuter + (long long)(batch & 7) * cInner;
  int row = blockIdx.x * 16 + (lane & 15);
  int col0 = blockIdx.y * 64 + (lane & 15);
  int koff = (lane >> 4) << 3;
  const unsigned short* ap = A + aoff + (long long)row * lda;
  const unsigned short* bp = B + boff + (long long)col0 * ldb;
  const long long bstep = (long long)16 * ldb;
  v8f acc0 = {}, acc1 = {}, acc2 = {}, acc3 = {};
  for (int k0 = 0; k0 < Kdim; k0 += 32) {
    FragBF a, b0, b1, b2, b3;
    load_frag(a, ap + k0 + koff);
    const unsigned short* bk = bp + k0 + koff;
    load_frag(b0, bk);
    load_frag(b1, bk + bstep);
    load_frag(b2, bk + 2 * bstep);
    load_frag(b3, bk + 3 * bstep);
    acc0 = wmma_bf16(a, b0, acc0);
    acc1 = wmma_bf16(a, b1, acc1);
    acc2 = wmma_bf16(a, b2, acc2);
    acc3 = wmma_bf16(a, b3, acc3);
  }
  int rbase = (lane >> 4) << 3;
  int m0 = blockIdx.x * 16;
  for (int v = 0; v < 8; ++v) {
    float* dst = C + coff + (long long)(m0 + rbase + v) * ldc + col0;
    dst[0]  = acc0[v] * scale;
    dst[16] = acc1[v] * scale;
    dst[32] = acc2[v] * scale;
    dst[48] = acc3[v] * scale;
  }
}

// row softmax with mask; f32 scores -> bf16 probs
__global__ void softmax_kernel(const float* __restrict__ scores,
                               const unsigned char* __restrict__ mask,
                               unsigned short* __restrict__ probs) {
  __shared__ float red[256];
  int q = blockIdx.x, bh = blockIdx.y;
  const float* row = scores + ((long long)bh * SEQ + q) * SEQ;
  const unsigned char* mrow = mask + (long long)q * SEQ;
  float v[4];
  float mx = -3.4e38f;
  for (int i = 0; i < 4; ++i) {
    int c = threadIdx.x + i * 256;
    float s = mrow[c] ? row[c] : -3.4e38f;
    v[i] = s;
    mx = fmaxf(mx, s);
  }
  red[threadIdx.x] = mx; __syncthreads();
  for (int off = 128; off > 0; off >>= 1) {
    if (threadIdx.x < off) red[threadIdx.x] = fmaxf(red[threadIdx.x], red[threadIdx.x + off]);
    __syncthreads();
  }
  mx = red[0]; __syncthreads();
  float sum = 0.f;
  for (int i = 0; i < 4; ++i) { v[i] = __expf(v[i] - mx); sum += v[i]; }
  red[threadIdx.x] = sum; __syncthreads();
  for (int off = 128; off > 0; off >>= 1) {
    if (threadIdx.x < off) red[threadIdx.x] += red[threadIdx.x + off];
    __syncthreads();
  }
  float inv = 1.f / red[0];
  unsigned short* prow = probs + ((long long)bh * SEQ + q) * SEQ;
  for (int i = 0; i < 4; ++i) {
    int c = threadIdx.x + i * 256;
    prow[c] = f2bf(v[i] * inv);
  }
}

// ---------------- host-side orchestration ----------------

static void run_sparse(hipStream_t stream,
                       const float* inF, const unsigned short* inB, int Din,
                       const float* W, int k, int Ncol, const unsigned short* Bbf,
                       const unsigned short* zrow,
                       float* outF, unsigned short* outB, int mode,
                       int* idx, float* wgt, int* counts, int* offsets, int* cursors,
                       int* table, int* entries, float* sel) {
  int maxTiles = NN + (T_TOK * k) / 16;
  zero_kernel<<<1, 64, 0, stream>>>(counts, NN);
  router_kernel<<<T_TOK, 64, 0, stream>>>(inF, W, Din, k, idx, wgt, counts);
  scan_kernel<<<1, 1, 0, stream>>>(counts, offsets, cursors, table, maxTiles);
  int total = T_TOK * k;
  scatter_kernel<<<(total + 255) / 256, 256, 0, stream>>>(idx, k, offsets, cursors, entries, total);
  moe_gemm_kernel<<<dim3(maxTiles, Ncol / 64), 32, 0, stream>>>(
      inB, Din, Bbf, Ncol, table, offsets, counts, entries, zrow, sel, k);
  combine_kernel<<<dim3(T_TOK, Ncol / 256), 256, 0, stream>>>(
      sel, wgt, k, Ncol, outF, outB, mode);
}

extern "C" void kernel_launch(void* const* d_in, const int* in_sizes, int n_in,
                              void* d_out, int out_size, void* d_ws, size_t ws_size,
                              hipStream_t stream) {
  (void)in_sizes; (void)n_in; (void)out_size; (void)ws_size;
  const float* x  = (const float*)d_in[0];
  const unsigned char* mask = (const unsigned char*)d_in[1];
  const float* cn = (const float*)d_in[2];
  const float* en = (const float*)d_in[3];
  const float* wq = (const float*)d_in[4];
  const float* wk = (const float*)d_in[5];
  const float* wv = (const float*)d_in[6];
  const float* wo = (const float*)d_in[7];
  float* out = (float*)d_out;

  char* ws = (char*)d_ws;
  auto alloc = [&](size_t bytes) -> char* {
    char* p = ws;
    ws += (bytes + 255) & ~(size_t)255;
    return p;
  };
  unsigned short* cnb    = (unsigned short*)alloc((size_t)NN * DM * RK * 2); // [n][r][d] bf16
  unsigned short* enb    = (unsigned short*)alloc((size_t)NN * DM * RK * 2); // [n][d][r] bf16
  unsigned short* xb     = (unsigned short*)alloc((size_t)T_TOK * DM * 2);
  unsigned short* qb     = (unsigned short*)alloc((size_t)T_TOK * RK * 2);
  unsigned short* kb     = (unsigned short*)alloc((size_t)T_TOK * RK * 2);
  unsigned short* vt     = (unsigned short*)alloc((size_t)BHN * DH * SEQ * 2);
  float*          scoresW= (float*)alloc((size_t)BHN * SEQ * SEQ * 4);
  unsigned short* probsW = (unsigned short*)alloc((size_t)BHN * SEQ * SEQ * 2);
  float*          attnF  = (float*)alloc((size_t)T_TOK * RK * 4);
  unsigned short* attnB  = (unsigned short*)alloc((size_t)T_TOK * RK * 2);
  float*          sel    = (float*)alloc((size_t)T_TOK * KC * RK * 4); // == T*KE*DM
  unsigned short* zrow   = (unsigned short*)alloc((size_t)DM * 2);     // zero dummy row
  int*   idx     = (int*)alloc((size_t)T_TOK * KC * 4);
  float* wgt     = (float*)alloc((size_t)T_TOK * KC * 4);
  int*   counts  = (int*)alloc(NN * 4);
  int*   offsets = (int*)alloc(NN * 4);
  int*   cursors = (int*)alloc(NN * 4);
  int*   table   = (int*)alloc((NN + (T_TOK * KC) / 16) * 4);
  int*   entries = (int*)alloc((size_t)T_TOK * KC * 4);

  // 1) cast neuron banks to bf16, K-contiguous (transposed) layout; cast x; zero dummy row
  transpose_cast_kernel<<<dim3(DM / 16, RK / 16, NN), dim3(16, 16), 0, stream>>>(cn, cnb, DM, RK);
  transpose_cast_kernel<<<dim3(RK / 16, DM / 16, NN), dim3(16, 16), 0, stream>>>(en, enb, RK, DM);
  cast_kernel<<<(T_TOK * DM) / 256, 256, 0, stream>>>(x, xb, T_TOK * DM);
  zero_kernel<<<8, 64, 0, stream>>>((int*)zrow, (DM * 2) / 4);

  // 2) sparse compressors Q/K/V  ([T,1024] -> [T,512])
  run_sparse(stream, x, xb, DM, wq, KC, RK, cnb, zrow, nullptr, qb, 1,
             idx, wgt, counts, offsets, cursors, table, entries, sel);
  run_sparse(stream, x, xb, DM, wk, KC, RK, cnb, zrow, nullptr, kb, 1,
             idx, wgt, counts, offsets, cursors, table, entries, sel);
  run_sparse(stream, x, xb, DM, wv, KC, RK, cnb, zrow, nullptr, vt, 2,
             idx, wgt, counts, offsets, cursors, table, entries, sel);

  // 3) attention: scores = Q K^T / sqrt(64), softmax(mask), out = P V
  gemm_bf16_kernel<<<dim3(SEQ / 16, SEQ / 64, BHN), 32, 0, stream>>>(
      qb, (long long)SEQ * RK, DH, RK,
      kb, (long long)SEQ * RK, DH, RK,
      scoresW, 8ll * SEQ * SEQ, (long long)SEQ * SEQ, SEQ,
      DH, 0.125f);
  softmax_kernel<<<dim3(SEQ, BHN), 256, 0, stream>>>(scoresW, mask, probsW);
  gemm_bf16_kernel<<<dim3(SEQ / 16, DH / 64, BHN), 32, 0, stream>>>(
      probsW, 8ll * SEQ * SEQ, (long long)SEQ * SEQ, SEQ,
      vt, (long long)NH * DH * SEQ, (long long)DH * SEQ, SEQ,
      attnF, (long long)SEQ * RK, DH, RK,
      SEQ, 1.0f);
  cast_kernel<<<(T_TOK * RK) / 256, 256, 0, stream>>>(attnF, attnB, T_TOK * RK);

  // 4) sparse expander ([T,512] -> [T,1024]) straight into d_out (f32)
  run_sparse(stream, attnF, attnB, RK, wo, KE, DM, enb, zrow, out, nullptr, 0,
             idx, wgt, counts, offsets, cursors, table, entries, sel);
}
